// AdjMatmulEncoder_85615878078606
// MI455X (gfx1250) — matrix operand order repaired and checked by reading, not compiled, not run on hardware
//
#include <hip/hip_runtime.h>
#include <hip/hip_bf16.h>
#include <stdint.h>

// ---------------------------------------------------------------------------
// AdjMatmulEncoder for MI455X (gfx1250, wave32, WMMA).
// All heavy GEMMs run as v_wmma_f32_16x16x32_bf16 with fp32 accumulation;
// softmax / layernorm / residual kept in fp32. State (151MB) ~fits in 192MB L2.
// Main GEMM staging uses the gfx1250 async global->LDS path when available.
// ---------------------------------------------------------------------------

#define NN 96
#define BS 8
#define HID 512
#define HEADS 8
#define MROWS (NN * NN * BS) /* 73728 */
#define RELD 128
#define VOCABN 100
#define LN_EPS 1e-5f

#if __has_builtin(__builtin_amdgcn_global_load_async_to_lds_b128) && \
    __has_builtin(__builtin_amdgcn_s_wait_asynccnt)
#define HAVE_ASYNC_LDS 1
#endif

typedef __attribute__((ext_vector_type(16))) __bf16 v16bf;
typedef __attribute__((ext_vector_type(8)))  float  v8f;
typedef __attribute__((ext_vector_type(4)))  unsigned int u32x4;

#ifdef HAVE_ASYNC_LDS
// builtin signature (from hipcc diagnostic): (v4i AS1*, v4i AS3*, imm, imm)
typedef __attribute__((ext_vector_type(4))) int i32x4v;
typedef __attribute__((address_space(1))) i32x4v* glob_v4p;
typedef __attribute__((address_space(3))) i32x4v* lds_v4p;
#endif

union FragAB { v16bf v; u32x4 q[2]; };
union FragC  { v8f v; float f[8]; };
union U4U16  { u32x4 q; unsigned short s[8]; };

__device__ __forceinline__ unsigned short f2bf(float f) {
    unsigned int u = __float_as_uint(f);
    u += 0x7FFFu + ((u >> 16) & 1u);   // round-to-nearest-even
    return (unsigned short)(u >> 16);
}
__device__ __forceinline__ float wred_max(float v) {
#pragma unroll
    for (int o = 16; o > 0; o >>= 1) v = fmaxf(v, __shfl_xor(v, o, 32));
    return v;
}
__device__ __forceinline__ float wred_sum(float v) {
#pragma unroll
    for (int o = 16; o > 0; o >>= 1) v += __shfl_xor(v, o, 32);
    return v;
}

// ---------------- rel_base = rel_embed[src_tokens] @ rel_proj_w^T + b -------
__global__ void rel_base_k(const int* __restrict__ src_tokens,
                           const float* __restrict__ rel_embed,
                           const float* __restrict__ w,   // (HID, RELD)
                           const float* __restrict__ b,
                           float* __restrict__ out) {     // (VOCABN, HID)
    int idx = blockIdx.x * blockDim.x + threadIdx.x;
    if (idx >= VOCABN * HID) return;
    int v = idx / HID, o = idx % HID;
    const float* e  = rel_embed + (size_t)src_tokens[v] * RELD;
    const float* wr = w + (size_t)o * RELD;
    float acc = 0.f;
#pragma unroll 4
    for (int d = 0; d < RELD; ++d) acc += e[d] * wr[d];
    out[idx] = acc + b[o];
}

// ---------------- fp32 -> bf16 bulk convert --------------------------------
__global__ void cvt_bf16_k(const float* __restrict__ src,
                           unsigned short* __restrict__ dst, int n) {
    for (int i = blockIdx.x * blockDim.x + threadIdx.x; i < n;
         i += gridDim.x * blockDim.x)
        dst[i] = f2bf(src[i]);
}

// ---------------- relation gather: state fp32 + bf16 copies ----------------
__global__ void gather_k(const int* __restrict__ rel_type,
                         const float* __restrict__ rel_base,
                         float* __restrict__ state,
                         unsigned short* __restrict__ state_bf,
                         unsigned short* __restrict__ rel_bf) {
    int m = blockIdx.x;                       // (i*96+t)*8+b
    int r = rel_type[m];
    const float* src = rel_base + (size_t)r * HID;
    size_t base = (size_t)m * HID;
    for (int h = threadIdx.x; h < HID; h += blockDim.x) {
        float v = src[h];
        unsigned short bv = f2bf(v);
        state[base + h] = v;
        state_bf[base + h] = bv;
        rel_bf[base + h] = bv;
    }
}

// ---------------- main GEMM: C = A(MxK,bf16) * W(NxK,bf16)^T, N=512 --------
// block tile 128x64, 8 waves (4x2), wave tile 32x32 (2x2 WMMA accums)
// mode 0: out_bf = (acc+bias)*alpha   (Q/K projections)
// mode 1: out_f  =  acc+bias          (final output projection)
// mode 2: out_f  =  resid + relu(acc+bias)   (path projection + residual)
__global__ __launch_bounds__(256) void gemm_k(
        const unsigned short* __restrict__ A,
        const unsigned short* __restrict__ W,
        const float* __restrict__ bias,
        unsigned short* __restrict__ outb,
        float* __restrict__ outf,
        const float* __restrict__ resid,
        int K, float alpha, int mode) {
    __shared__ unsigned short As[128][40];
    __shared__ unsigned short Ws[64][40];
    const int tid = threadIdx.x;
    const int lane = tid & 31, wave = tid >> 5;
    const int wm = wave >> 1, wn = wave & 1;
    const int m0 = blockIdx.x * 128;
    const int n0 = blockIdx.y * 64;

    FragC acc[2][2];
#pragma unroll
    for (int i = 0; i < 2; ++i)
#pragma unroll
        for (int j = 0; j < 2; ++j)
#pragma unroll
            for (int r = 0; r < 8; ++r) acc[i][j].f[r] = 0.f;

    for (int k0 = 0; k0 < K; k0 += 32) {
#ifdef HAVE_ASYNC_LDS
        {   // async global->LDS staging (no VGPR round-trip, ASYNCcnt tracked)
            int row = tid >> 1, half = tid & 1;
            const unsigned short* g = A + (size_t)(m0 + row) * K + k0 + half * 16;
            __builtin_amdgcn_global_load_async_to_lds_b128(
                (glob_v4p)g, (lds_v4p)&As[row][half * 16], 0, 0);
            __builtin_amdgcn_global_load_async_to_lds_b128(
                (glob_v4p)(g + 8), (lds_v4p)&As[row][half * 16 + 8], 0, 0);
            if (tid < 128) {
                const unsigned short* gw =
                    W + (size_t)(n0 + row) * K + k0 + half * 16;
                __builtin_amdgcn_global_load_async_to_lds_b128(
                    (glob_v4p)gw, (lds_v4p)&Ws[row][half * 16], 0, 0);
                __builtin_amdgcn_global_load_async_to_lds_b128(
                    (glob_v4p)(gw + 8), (lds_v4p)&Ws[row][half * 16 + 8], 0, 0);
            }
            __builtin_amdgcn_s_wait_asynccnt(0);
        }
#else
        {   // stage A tile 128x32 (each thread: one 32B row segment)
            int row = tid >> 1, half = tid & 1;
            const unsigned short* g = A + (size_t)(m0 + row) * K + k0 + half * 16;
            u32x4 d0 = *(const u32x4*)g;
            u32x4 d1 = *(const u32x4*)(g + 8);
            *(u32x4*)&As[row][half * 16]     = d0;
            *(u32x4*)&As[row][half * 16 + 8] = d1;
        }
        if (tid < 128) {  // stage W tile 64x32
            int row = tid >> 1, half = tid & 1;
            const unsigned short* g = W + (size_t)(n0 + row) * K + k0 + half * 16;
            u32x4 d0 = *(const u32x4*)g;
            u32x4 d1 = *(const u32x4*)(g + 8);
            *(u32x4*)&Ws[row][half * 16]     = d0;
            *(u32x4*)&Ws[row][half * 16 + 8] = d1;
        }
#endif
        __syncthreads();
        FragAB a[2], bf[2];
#pragma unroll
        for (int t = 0; t < 2; ++t) {
            // A frag: lane<16 -> row m, K {0..7,16..23}; lane>=16 -> K {8..15,24..31}
            int row = wm * 32 + t * 16 + (lane & 15);
            int ca  = (lane < 16) ? 0 : 8;
            a[t].q[0] = *(const u32x4*)&As[row][ca];
            a[t].q[1] = *(const u32x4*)&As[row][ca + 16];
            // B frag: lane<16 -> col n, K {0..7,16..23}; lane>=16 -> K {8..15,24..31}
            int nrow = wn * 32 + t * 16 + (lane & 15);
            int cb   = (lane < 16) ? 0 : 8;
            bf[t].q[0] = *(const u32x4*)&Ws[nrow][cb];
            bf[t].q[1] = *(const u32x4*)&Ws[nrow][cb + 16];
        }
#pragma unroll
        for (int i = 0; i < 2; ++i)
#pragma unroll
            for (int j = 0; j < 2; ++j)
                acc[i][j].v = __builtin_amdgcn_wmma_f32_16x16x32_bf16(
                    false, a[i].v, false, bf[j].v, (short)0, acc[i][j].v,
                    false, false);
        __syncthreads();
    }
    const int mb = m0 + wm * 32 + ((lane < 16) ? 0 : 8);
    const int nb = n0 + wn * 32 + (lane & 15);
#pragma unroll
    for (int i = 0; i < 2; ++i) {
#pragma unroll
        for (int j = 0; j < 2; ++j) {
            int n = nb + j * 16;
            float bbv = bias[n];
#pragma unroll
            for (int r = 0; r < 8; ++r) {
                int m = mb + i * 16 + r;
                size_t o = (size_t)m * HID + n;
                float vv = acc[i][j].f[r] + bbv;
                if (mode == 0) {
                    outb[o] = f2bf(vv * alpha);
                } else if (mode == 1) {
                    outf[o] = vv;
                } else {
                    outf[o] = resid[o] + (vv > 0.f ? vv : 0.f);
                }
            }
        }
    }
}

// ---------------- fused scores + masked softmax + head-mean ----------------
// one block per (i,b); scores per head via WMMA into LDS; mean kept in regs.
__global__ __launch_bounds__(256) void attn_k(
        const unsigned short* __restrict__ q,
        const unsigned short* __restrict__ kmat,
        const int* __restrict__ rel_type,
        unsigned short* __restrict__ attn) {
    __shared__ float sc[96][100];
    __shared__ int smask[96];
    const int ib = blockIdx.x;
    const int i = ib >> 3, b = ib & 7;
    const int tid = threadIdx.x, lane = tid & 31, wave = tid >> 5;
    if (tid < 96) {
        int s = tid;
        smask[s] = (rel_type[(size_t)(i * 96 + s) * 8 + b] == 0 && s != i);
    }
    __syncthreads();

    float accv[12][3];   // rows t = wave+8*rr, cols s = lane+32*j (stable owner)
#pragma unroll
    for (int rr = 0; rr < 12; ++rr)
#pragma unroll
        for (int j = 0; j < 3; ++j) accv[rr][j] = 0.f;

    for (int head = 0; head < HEADS; ++head) {
        for (int tile = wave; tile < 36; tile += 8) {      // 6x6 t,s tiles
            int tm = tile / 6, tn = tile % 6;
            FragC c;
#pragma unroll
            for (int r = 0; r < 8; ++r) c.f[r] = 0.f;
#pragma unroll
            for (int dstep = 0; dstep < 2; ++dstep) {      // d = 0..63
                int d0 = dstep * 32;
                FragAB a, bb;
                int t = tm * 16 + (lane & 15);
                size_t qb = ((size_t)(i * 96 + t) * 8 + b) * 512 + head * 64 +
                            d0 + ((lane < 16) ? 0 : 8);
                a.q[0] = *(const u32x4*)(q + qb);
                a.q[1] = *(const u32x4*)(q + qb + 16);
                int s = tn * 16 + (lane & 15);
                size_t kb = ((size_t)(i * 96 + s) * 8 + b) * 512 + head * 64 +
                            d0 + ((lane < 16) ? 0 : 8);
                bb.q[0] = *(const u32x4*)(kmat + kb);
                bb.q[1] = *(const u32x4*)(kmat + kb + 16);
                c.v = __builtin_amdgcn_wmma_f32_16x16x32_bf16(
                    false, a.v, false, bb.v, (short)0, c.v, false, false);
            }
            int scs = tn * 16 + (lane & 15);
            int rb = tm * 16 + ((lane < 16) ? 0 : 8);
#pragma unroll
            for (int r = 0; r < 8; ++r) sc[rb + r][scs] = c.f[r];
        }
        __syncthreads();
#pragma unroll 1
        for (int rr = 0; rr < 12; ++rr) {                  // masked softmax rows
            int t = wave + rr * 8;
            float v0[3];
#pragma unroll
            for (int j = 0; j < 3; ++j) {
                int s = lane + 32 * j;
                v0[j] = smask[s] ? -3.0e38f : sc[t][s];
            }
            float mx = wred_max(fmaxf(fmaxf(v0[0], v0[1]), v0[2]));
            float p[3], sum = 0.f;
#pragma unroll
            for (int j = 0; j < 3; ++j) { p[j] = __expf(v0[j] - mx); sum += p[j]; }
            sum = wred_sum(sum);
            float inv = 1.0f / sum;
#pragma unroll
            for (int j = 0; j < 3; ++j) accv[rr][j] += p[j] * inv;
        }
        __syncthreads();
    }
    size_t base = (size_t)ib * 96 * 96;                    // (i,b,t,s)
#pragma unroll
    for (int rr = 0; rr < 12; ++rr) {
        int t = wave + rr * 8;
#pragma unroll
        for (int j = 0; j < 3; ++j) {
            int s = lane + 32 * j;
            attn[base + (size_t)t * 96 + s] = f2bf(accv[rr][j] * 0.125f);
        }
    }
}

// ---------------- batched strided GEMM (x / rel_mix einsums) ---------------
// M=96,K=96,N=512 per batch (p,b). B operand is N-contiguous per K row,
// so it is transposed through LDS into Bs[n][k] for the WMMA B layout.
// block: one (p,b) batch + one 16-row m-tile; 8 waves x 64 cols.
__global__ __launch_bounds__(256) void apply_k(
        const unsigned short* __restrict__ Ab,
        const unsigned short* __restrict__ Bb,
        unsigned short* __restrict__ Ob,
        int A_pOff, int A_bOff, int A_mStr,
        int B_pOff, int B_bOff, int B_kStr,
        int O_pOff, int O_bOff, int O_mStr, int O_extra) {
    __shared__ unsigned short Bs[512][40];
    const int tid = threadIdx.x, lane = tid & 31, wave = tid >> 5;
    const int pb = blockIdx.x;
    const int p = pb >> 3, b = pb & 7;
    const int m0 = blockIdx.y * 16;
    const unsigned short* Ap = Ab + (size_t)p * A_pOff + (size_t)b * A_bOff;
    const unsigned short* Bp = Bb + (size_t)p * B_pOff + (size_t)b * B_bOff;
    unsigned short* Op = Ob + (size_t)p * O_pOff + (size_t)b * O_bOff + O_extra;
    const int n0w = wave * 64;

    FragC acc[4];
#pragma unroll
    for (int j = 0; j < 4; ++j)
#pragma unroll
        for (int r = 0; r < 8; ++r) acc[j].f[r] = 0.f;

    for (int k0 = 0; k0 < 96; k0 += 32) {
        {   // stage+transpose B tile 32(k) x 512(n) -> Bs[n][k]
            int k  = tid >> 3;
            int nc = tid & 7;
            const unsigned short* gk = Bp + (size_t)(k0 + k) * B_kStr;
#pragma unroll
            for (int c = 0; c < 8; ++c) {
                int n = (nc + c * 8) * 8;
                U4U16 u; u.q = *(const u32x4*)(gk + n);
#pragma unroll
                for (int j = 0; j < 8; ++j) Bs[n + j][k] = u.s[j];
            }
        }
        __syncthreads();
        FragAB a;
        {
            int m = m0 + (lane & 15);
            size_t ao = (size_t)m * A_mStr + k0 + ((lane < 16) ? 0 : 8);
            a.q[0] = *(const u32x4*)(Ap + ao);
            a.q[1] = *(const u32x4*)(Ap + ao + 16);
        }
#pragma unroll
        for (int j = 0; j < 4; ++j) {
            FragAB bf;
            int n  = n0w + j * 16 + (lane & 15);
            int cb = (lane < 16) ? 0 : 8;
            bf.q[0] = *(const u32x4*)&Bs[n][cb];
            bf.q[1] = *(const u32x4*)&Bs[n][cb + 16];
            acc[j].v = __builtin_amdgcn_wmma_f32_16x16x32_bf16(
                false, a.v, false, bf.v, (short)0, acc[j].v, false, false);
        }
        __syncthreads();
    }
    const int mb = m0 + ((lane < 16) ? 0 : 8);
#pragma unroll
    for (int j = 0; j < 4; ++j) {
        int n = n0w + j * 16 + (lane & 15);
#pragma unroll
        for (int r = 0; r < 8; ++r) {
            int m = mb + r;
            Op[(size_t)m * O_mStr + n] = f2bf(acc[j].f[r]);
        }
    }
}

// ---------------- LayerNorm + re-quantize to bf16 --------------------------
__global__ __launch_bounds__(256) void ln_k(
        const float* __restrict__ res,
        const float* __restrict__ g,
        const float* __restrict__ beta,
        float* __restrict__ state,
        unsigned short* __restrict__ state_bf) {
    const int tid = threadIdx.x, lane = tid & 31, wave = tid >> 5;
    const int row = blockIdx.x * 8 + wave;
    const float* rp = res + (size_t)row * HID;
    float vals[16], sum = 0.f, sq = 0.f;
#pragma unroll
    for (int j = 0; j < 16; ++j) {
        float v = rp[lane + 32 * j];
        vals[j] = v; sum += v; sq += v * v;
    }
    sum = wred_sum(sum); sq = wred_sum(sq);
    float mean = sum * (1.0f / HID);
    float var  = sq * (1.0f / HID) - mean * mean;
    float rstd = rsqrtf(var + LN_EPS);
    size_t base = (size_t)row * HID;
#pragma unroll
    for (int j = 0; j < 16; ++j) {
        int h = lane + 32 * j;
        float o = (vals[j] - mean) * rstd * g[h] + beta[h];
        state[base + h] = o;
        state_bf[base + h] = f2bf(o);
    }
}

// ---------------------------------------------------------------------------
extern "C" void kernel_launch(void* const* d_in, const int* in_sizes, int n_in,
                              void* d_out, int out_size, void* d_ws, size_t ws_size,
                              hipStream_t stream) {
    const int*   src_tokens = (const int*)d_in[0];
    const int*   rel_type   = (const int*)d_in[2];
    const float* rel_embed  = (const float*)d_in[3];
    const float* rel_proj_w = (const float*)d_in[4];
    const float* rel_proj_b = (const float*)d_in[5];
    const float* wq    = (const float*)d_in[6];
    const float* bq    = (const float*)d_in[7];
    const float* wk    = (const float*)d_in[8];
    const float* bk    = (const float*)d_in[9];
    const float* pathw = (const float*)d_in[10];
    const float* pathb = (const float*)d_in[11];
    const float* lng   = (const float*)d_in[12];
    const float* lnb   = (const float*)d_in[13];
    const float* outw  = (const float*)d_in[14];
    const float* outb  = (const float*)d_in[15];

    char* ws = (char*)d_ws;
    size_t off = 0;
    auto alloc = [&](size_t bytes) -> void* {
        void* pp = ws + off;
        off = (off + bytes + 255) & ~(size_t)255;
        return pp;
    };
    float* state            = (float*)alloc((size_t)MROWS * HID * 4);
    float* res              = (float*)alloc((size_t)MROWS * HID * 4);
    unsigned short* state_bf = (unsigned short*)alloc((size_t)MROWS * HID * 2);
    unsigned short* rel_bf   = (unsigned short*)alloc((size_t)MROWS * HID * 2);
    unsigned short* q_bf     = (unsigned short*)alloc((size_t)MROWS * HID * 2);
    unsigned short* k_bf     = (unsigned short*)alloc((size_t)MROWS * HID * 2);
    unsigned short* attn_bf  = (unsigned short*)alloc((size_t)NN * BS * NN * NN * 2);
    unsigned short* xcat_bf  = (unsigned short*)alloc((size_t)MROWS * 2 * HID * 2);
    float* rel_base          = (float*)alloc((size_t)VOCABN * HID * 4);
    unsigned short* wq_bf = (unsigned short*)alloc((size_t)512 * 512 * 2);
    unsigned short* wk_bf = (unsigned short*)alloc((size_t)512 * 512 * 2);
    unsigned short* pw_bf = (unsigned short*)alloc((size_t)512 * 1024 * 2);
    unsigned short* ow_bf = (unsigned short*)alloc((size_t)512 * 512 * 2);

    rel_base_k<<<(VOCABN * HID + 255) / 256, 256, 0, stream>>>(
        src_tokens, rel_embed, rel_proj_w, rel_proj_b, rel_base);
    cvt_bf16_k<<<512, 256, 0, stream>>>(wq, wq_bf, 512 * 512);
    cvt_bf16_k<<<512, 256, 0, stream>>>(wk, wk_bf, 512 * 512);
    cvt_bf16_k<<<512, 256, 0, stream>>>(pathw, pw_bf, 512 * 1024);
    cvt_bf16_k<<<512, 256, 0, stream>>>(outw, ow_bf, 512 * 512);
    gather_k<<<MROWS, 256, 0, stream>>>(rel_type, rel_base, state, state_bf, rel_bf);

    dim3 ggrid(MROWS / 128, HID / 64);
    for (int layer = 0; layer < 2; ++layer) {
        // q = (state@wq^T + bq) * 1/sqrt(64); k = state@wk^T + bk  (bf16 out)
        gemm_k<<<ggrid, 256, 0, stream>>>(state_bf, wq_bf, bq, q_bf, nullptr,
                                          nullptr, HID, 0.125f, 0);
        gemm_k<<<ggrid, 256, 0, stream>>>(state_bf, wk_bf, bk, k_bf, nullptr,
                                          nullptr, HID, 1.0f, 0);
        attn_k<<<NN * BS, 256, 0, stream>>>(q_bf, k_bf, rel_type, attn_bf);
        // x[i,t,b,h] = sum_s attn[i,b,t,s] * state[i,s,b,h]  -> xcat[:, :512]
        apply_k<<<dim3(NN * BS, NN / 16), 256, 0, stream>>>(
            attn_bf, state_bf, xcat_bf,
            /*A_pOff*/ 8 * 96 * 96, /*A_bOff*/ 96 * 96, /*A_mStr*/ 96,
            /*B_pOff*/ 96 * 8 * 512, /*B_bOff*/ 512, /*B_kStr*/ 8 * 512,
            /*O_pOff*/ 96 * 8 * 1024, /*O_bOff*/ 1024, /*O_mStr*/ 8 * 1024,
            /*O_extra*/ 0);
        // rel_mix[i,t,b,h] = sum_s attn[i,b,t,s] * relation[s,t,b,h] -> xcat[:,512:]
        apply_k<<<dim3(NN * BS, NN / 16), 256, 0, stream>>>(
            attn_bf, rel_bf, xcat_bf,
            /*A_pOff*/ 96, /*A_bOff*/ 96 * 96, /*A_mStr*/ 8 * 96 * 96,
            /*B_pOff*/ 8 * 512, /*B_bOff*/ 512, /*B_kStr*/ 96 * 8 * 512,
            /*O_pOff*/ 8 * 1024, /*O_bOff*/ 1024, /*O_mStr*/ 96 * 8 * 1024,
            /*O_extra*/ 512);
        // res = state + relu(xcat @ path_w^T + path_b)
        gemm_k<<<ggrid, 256, 0, stream>>>(xcat_bf, pw_bf, pathb, nullptr, res,
                                          state, 2 * HID, 1.0f, 2);
        ln_k<<<MROWS / 8, 256, 0, stream>>>(res, lng, lnb, state, state_bf);
    }
    // out = state @ out_w^T + out_b  (fp32)
    gemm_k<<<ggrid, 256, 0, stream>>>(state_bf, ow_bf, outb, nullptr,
                                      (float*)d_out, nullptr, HID, 1.0f, 1);
}
